// SudokuGNN_35545149342050
// MI455X (gfx1250) — matrix operand order repaired and checked by reading, not compiled, not run on hardware
//
#include <hip/hip_runtime.h>
#include <hip/hip_bf16.h>
#include <hip/hip_fp16.h>

typedef __attribute__((ext_vector_type(16))) _Float16 v16h;
typedef __attribute__((ext_vector_type(2)))  __fp16   v2fp16;
typedef __attribute__((ext_vector_type(8)))  float    v8f;

// ---------------- problem constants ----------------
#define NGRAPH 2048
#define HID    128
// padded node count per graph: 81 -> 96 (6 tiles of 16)
// LDS strides (f16 units, all multiples of 8 so every fragment chunk is 16B aligned)
#define SA  104   // Ahat [96][SA]
#define SX   32   // X    [96][SX]  (K padded 10->32)
#define SZ  104   // zT   [128][SZ] (transposed Z: row=feat, col=node)
#define SH  136   // hRow [96][SH]  (row=node, col=feat)

// LDS layout offsets (f16 units; all multiples of 8)
#define OFF_A   0
#define OFF_X   (OFF_A  + 96*SA)          // 9984
#define OFF_W1  (OFF_X  + 96*SX)          // 13056
#define OFF_W2  (OFF_W1 + 128*32)         // 17152
#define OFF_ZT  (OFF_W2 + 128*128)        // 33536
#define OFF_H   (OFF_ZT + 128*SZ)         // 46848
#define F16_TOT (OFF_H  + 96*SH)          // 59904 f16 = 119808 B
#define BYTE_POOLED (F16_TOT*2)           // 119808
#define BYTE_B1     (BYTE_POOLED + 512)   // 120320
#define BYTE_B2     (BYTE_B1 + 512)       // 120832
#define LDS_BYTES   (BYTE_B2 + 512)       // 121344

// output segment offsets (tuple concatenated flat): [2048,4][2048,81][2048,9][2048,10]
#define OUT_A 0
#define OUT_C (NGRAPH*4)              // 8192
#define OUT_N (OUT_C + NGRAPH*81)     // 174080
#define OUT_T (OUT_N + NGRAPH*9)      // 192512

// ---------------- fragment helpers (all 16B-aligned b128 LDS ops) ----------------
// A fragment (16x32 f16 from row-major [m][k] buffer):
// lane m=lane&15, half=lane>>4; 16 contiguous k = half*16 + 0..15
// caller passes rb = base + row*stride + kbase + half*16  (f16 units, multiple of 8)
__device__ __forceinline__ v16h ldsAfrag(const char* smem, int rb) {
  const uint4* p = (const uint4*)(smem + rb * 2);
  union { v16h v; uint4 q[2]; } f;
  f.q[0] = p[0];           // bytes [0,16)   -> K 0..7  (local)
  f.q[1] = p[1];           // bytes [16,32)  -> K 8..15 (local)
  return f.v;
}

// B fragment (32x16 f16 from [n][k] buffer): lane n=lane&15; lanes 0-15 K=0..15,
// lanes 16-31 K=16..31, 16 contiguous f16 = 32 bytes.
// caller passes rb = base + n_row*stride + kbase + half*16  (multiple of 8)
__device__ __forceinline__ v16h ldsBfrag(const char* smem, int rb) {
  const uint4* p = (const uint4*)(smem + rb * 2);
  union { v16h v; uint4 q[2]; } f;
  f.q[0] = p[0];
  f.q[1] = p[1];
  return f.v;
}

__device__ __forceinline__ v8f wmma16(v16h a, v16h b, v8f c) {
  return __builtin_amdgcn_wmma_f32_16x16x32_f16(false, a, false, b, (short)0, c, false, false);
}

__device__ __forceinline__ uint32_t pk2(float a, float b) {
#if __has_builtin(__builtin_amdgcn_cvt_pkrtz)
  union { v2fp16 h; uint32_t u; } p;
  p.h = __builtin_amdgcn_cvt_pkrtz(a, b);
  return p.u;
#else
  union { _Float16 h[2]; uint32_t u; } p;
  p.h[0] = (_Float16)a; p.h[1] = (_Float16)b;
  return p.u;
#endif
}

// Contiguous D store: lane holds col N, rows M contiguous -> one ds_store_b128
// writes buffer layout [N][M]. idx16 must be a multiple of 8 (guaranteed).
__device__ __forceinline__ void storeD16(char* smem, int idx16, v8f c) {
  uint4 q;
  q.x = pk2(c[0], c[1]); q.y = pk2(c[2], c[3]);
  q.z = pk2(c[4], c[5]); q.w = pk2(c[6], c[7]);
  *(uint4*)(smem + idx16 * 2) = q;
}

// ---------------- kernel 0: weight prep (fp32 -> f16, transposed [n][k]) ------
__global__ void sudoku_prep_weights(const float* __restrict__ W1,
                                    const float* __restrict__ W2,
                                    _Float16* __restrict__ w1t,
                                    _Float16* __restrict__ w2t) {
  const int tid = blockIdx.x * blockDim.x + threadIdx.x;
  const int nt  = gridDim.x * blockDim.x;
  for (int i = tid; i < 128 * 32; i += nt) {           // W1: [10,128] -> [128][32] (K pad->32)
    const int j = i >> 5, k = i & 31;
    w1t[i] = (k < 10) ? (_Float16)W1[k * 128 + j] : (_Float16)0.0f;
  }
  for (int i = tid; i < 128 * 128; i += nt) {          // W2: [128,128] -> [128][128] transposed
    const int j = i >> 7, k = i & 127;
    w2t[i] = (_Float16)W2[k * 128 + j];
  }
}

// ---------------- kernel 1: one workgroup (8 waves) per graph ----------------
__global__ __launch_bounds__(256) void sudoku_gnn_kernel(
    const float* __restrict__ x,
    const float* __restrict__ b1g, const float* __restrict__ b2g,
    const _Float16* __restrict__ w1t, const _Float16* __restrict__ w2t,
    const float* __restrict__ Wa, const float* __restrict__ ba,
    const float* __restrict__ Wc, const float* __restrict__ bc,
    const float* __restrict__ Wn, const float* __restrict__ bn,
    const float* __restrict__ Wt, const float* __restrict__ bt,
    float* __restrict__ out)
{
  extern __shared__ char smem[];
  _Float16* lds   = (_Float16*)smem;
  uint32_t* lds32 = (uint32_t*)smem;
  float* pooled = (float*)(smem + BYTE_POOLED);
  float* b1s    = (float*)(smem + BYTE_B1);
  float* b2s    = (float*)(smem + BYTE_B2);

  const int g    = blockIdx.x;
  const int tid  = threadIdx.x;
  const int wave = tid >> 5;          // 0..7 : owns one 16-wide column tile per phase
  const int lane = tid & 31;
  const int half = lane >> 4;
  const int l16  = lane & 15;

  // --- build adjacency Ahat (0/1 f16; symmetric; self-loop on diagonal; pads 0) ---
  for (int i = tid; i < 96 * 96; i += 256) {
    const int r = i / 96, c = i % 96;
    _Float16 v = (_Float16)0.0f;
    if (r < 81 && c < 81) {
      const int ri = r / 9, ci = r % 9, rj = c / 9, cj = c % 9;
      if (ri == rj || ci == cj || ((ri / 3 == rj / 3) && (ci / 3 == cj / 3)))
        v = (_Float16)1.0f;
    }
    lds[OFF_A + r * SA + c] = v;
  }
  // --- stage X tile [96][32] (rows>=81, cols>=10 zero) ---
  for (int i = tid; i < 96 * 32; i += 256) {
    const int r = i >> 5, c = i & 31;
    float v = 0.0f;
    if (r < 81 && c < 10) v = x[(g * 81 + r) * 10 + c];
    lds[OFF_X + r * SX + c] = (_Float16)v;
  }
  // --- stage pre-converted f16 weights ---
  for (int i = tid; i < (128 * 32) / 2;  i += 256) lds32[(OFF_W1 >> 1) + i] = ((const uint32_t*)w1t)[i];
  for (int i = tid; i < (128 * 128) / 2; i += 256) lds32[(OFF_W2 >> 1) + i] = ((const uint32_t*)w2t)[i];
  if (tid < 128) { b1s[tid] = b1g[tid]; b2s[tid] = b2g[tid]; pooled[tid] = 0.0f; }
  __syncthreads();

  // ===== Phase 1: Z1 = X @ W1  (M=node, N=feat) -> contiguous store = zT[feat][node] =====
  {
    const int ntF = wave;                                       // feat tile 0..7
    const v16h b = ldsBfrag(smem, OFF_W1 + (ntF * 16 + l16) * 32 + half * 16);
    for (int mt = 0; mt < 6; ++mt) {
      v16h a = ldsAfrag(smem, OFF_X + (mt * 16 + l16) * SX + half * 16);
      v8f c = {};
      c = wmma16(a, b, c);
      storeD16(smem, OFF_ZT + (ntF * 16 + l16) * SZ + mt * 16 + half * 8, c);
    }
  }
  __syncthreads();

  // ===== Phase 2: H1^T = Z1^T @ Ahat (Ahat symmetric)  (M=feat, N=node)
  //        relu((1/21)*acc + b1[featM]) -> contiguous store = hRow[node][feat] =====
  {
    const int mtF = wave;                                       // feat tile 0..7
    const int arow = OFF_ZT + (mtF * 16 + l16) * SZ + half * 16;
    const v16h a0 = ldsAfrag(smem, arow +  0);
    const v16h a1 = ldsAfrag(smem, arow + 32);
    const v16h a2 = ldsAfrag(smem, arow + 64);
    float bias[8];
#pragma unroll
    for (int r = 0; r < 8; ++r) bias[r] = b1s[mtF * 16 + half * 8 + r];
    for (int nt = 0; nt < 6; ++nt) {                            // node-out tiles
      const int brow = OFF_A + (nt * 16 + l16) * SA + half * 16;
      v8f c = {};
      c = wmma16(a0, ldsBfrag(smem, brow +  0), c);
      c = wmma16(a1, ldsBfrag(smem, brow + 32), c);
      c = wmma16(a2, ldsBfrag(smem, brow + 64), c);
#pragma unroll
      for (int r = 0; r < 8; ++r) {
        float v = c[r] * (1.0f / 21.0f) + bias[r];
        c[r] = v > 0.0f ? v : 0.0f;
      }
      storeD16(smem, OFF_H + (nt * 16 + l16) * SH + mtF * 16 + half * 8, c);
    }
  }
  __syncthreads();

  // ===== Phase 3: Z2 = H1 @ W2  (M=node, N=featOut) -> contiguous store = zT[feat][node] =====
  {
    const int ntF = wave;                                       // featOut tile 0..7
    const int brow = OFF_W2 + (ntF * 16 + l16) * 128 + half * 16;
    const v16h b0 = ldsBfrag(smem, brow +  0);
    const v16h b1f = ldsBfrag(smem, brow + 32);
    const v16h b2f = ldsBfrag(smem, brow + 64);
    const v16h b3f = ldsBfrag(smem, brow + 96);
    for (int mt = 0; mt < 6; ++mt) {
      const int arow = OFF_H + (mt * 16 + l16) * SH + half * 16;
      v8f c = {};
      c = wmma16(ldsAfrag(smem, arow +  0), b0,  c);
      c = wmma16(ldsAfrag(smem, arow + 32), b1f, c);
      c = wmma16(ldsAfrag(smem, arow + 64), b2f, c);
      c = wmma16(ldsAfrag(smem, arow + 96), b3f, c);
      storeD16(smem, OFF_ZT + (ntF * 16 + l16) * SZ + mt * 16 + half * 8, c);
    }
  }
  __syncthreads();

  // ===== Phase 4: H2 = Ahat @ Z2  (M=node, N=feat); relu + masked mean-pool fold =====
  {
    const int ntF = wave;                                       // feat tile 0..7
    const int brow = OFF_ZT + (ntF * 16 + l16) * SZ + half * 16;
    const v16h b0 = ldsBfrag(smem, brow +  0);
    const v16h b1f = ldsBfrag(smem, brow + 32);
    const v16h b2f = ldsBfrag(smem, brow + 64);
    const float bias = b2s[ntF * 16 + l16];
    float acc = 0.0f;
    for (int mt = 0; mt < 6; ++mt) {
      const int arow = OFF_A + (mt * 16 + l16) * SA + half * 16;
      v8f c = {};
      c = wmma16(ldsAfrag(smem, arow +  0), b0,  c);
      c = wmma16(ldsAfrag(smem, arow + 32), b1f, c);
      c = wmma16(ldsAfrag(smem, arow + 64), b2f, c);
#pragma unroll
      for (int r = 0; r < 8; ++r) {
        const int m = mt * 16 + half * 8 + r;                   // node row
        float v = c[r] * (1.0f / 21.0f) + bias;
        v = v > 0.0f ? v : 0.0f;
        if (m < 81) acc += v;                                   // mask pad rows
      }
    }
    atomicAdd(&pooled[ntF * 16 + l16], acc);                    // ds_add_f32
  }
  __syncthreads();

  if (tid < 128) pooled[tid] *= (1.0f / 81.0f);                 // mean over 81 nodes
  __syncthreads();

  // ===== Phase 5: four linear heads (104 outputs per graph) =====
  if (tid < 104) {
    const float* W; const float* bb; int ncol, j; long obase;
    if (tid < 4)       { W = Wa; bb = ba; ncol = 4;  j = tid;      obase = OUT_A + (long)g * 4;  }
    else if (tid < 85) { W = Wc; bb = bc; ncol = 81; j = tid - 4;  obase = OUT_C + (long)g * 81; }
    else if (tid < 94) { W = Wn; bb = bn; ncol = 9;  j = tid - 85; obase = OUT_N + (long)g * 9;  }
    else               { W = Wt; bb = bt; ncol = 10; j = tid - 94; obase = OUT_T + (long)g * 10; }
    float acc = bb[j];
#pragma unroll 8
    for (int k = 0; k < 128; ++k) acc += pooled[k] * W[k * ncol + j];
    out[obase + j] = acc;
  }
}

// ---------------- host launcher ----------------
extern "C" void kernel_launch(void* const* d_in, const int* in_sizes, int n_in,
                              void* d_out, int out_size, void* d_ws, size_t ws_size,
                              hipStream_t stream) {
  (void)in_sizes; (void)n_in; (void)out_size; (void)ws_size;
  // setup_inputs order:
  // 0:x 1:edge_index 2:batch 3:W1 4:b1 5:W2 6:b2 7:Wa 8:ba 9:Wc 10:bc 11:Wn 12:bn 13:Wt 14:bt
  const float* x  = (const float*)d_in[0];
  const float* W1 = (const float*)d_in[3];
  const float* b1 = (const float*)d_in[4];
  const float* W2 = (const float*)d_in[5];
  const float* b2 = (const float*)d_in[6];
  const float* Wa = (const float*)d_in[7];
  const float* ba = (const float*)d_in[8];
  const float* Wc = (const float*)d_in[9];
  const float* bc = (const float*)d_in[10];
  const float* Wn = (const float*)d_in[11];
  const float* bn = (const float*)d_in[12];
  const float* Wt = (const float*)d_in[13];
  const float* bt = (const float*)d_in[14];
  float* out = (float*)d_out;

  _Float16* w1t = (_Float16*)d_ws;            // 128*32 f16
  _Float16* w2t = w1t + 128 * 32;             // 128*128 f16

  (void)hipFuncSetAttribute((const void*)sudoku_gnn_kernel,
                            hipFuncAttributeMaxDynamicSharedMemorySize, LDS_BYTES);

  sudoku_prep_weights<<<32, 256, 0, stream>>>(W1, W2, w1t, w2t);
  sudoku_gnn_kernel<<<NGRAPH, 256, LDS_BYTES, stream>>>(
      x, b1, b2, w1t, w2t, Wa, ba, Wc, bc, Wn, bn, Wt, bt, out);
}
